// LNN_16234976379185
// MI455X (gfx1250) — matrix operand order repaired and checked by reading, not compiled, not run on hardware
//
#include <hip/hip_runtime.h>
#include <cstdint>
#include <cstddef>

typedef __attribute__((ext_vector_type(16))) __bf16 v16bf;
typedef __attribute__((ext_vector_type(8)))  float  v8f;

#define DEVFN static __device__ __forceinline__

union FragU {
    uint4  q[2];
    v16bf  v;
};

DEVFN unsigned short f2bf(float f) {
    uint32_t x = __float_as_uint(f);
    uint32_t lsb = (x >> 16) & 1u;
    x += 0x7FFFu + lsb;                 // round-to-nearest-even
    return (unsigned short)(x >> 16);
}

// A fragment, 16x32 bf16 tile of row-major [M,K] matrix at (row0, k0).
// ISA 7.12.2: lanes 0-15 hold M=lane, K chunks {0..7,16..23};
//             lanes 16-31 hold M=lane-16, K chunks {8..15,24..31}.
DEVFN v16bf load_fragA(const unsigned short* __restrict__ base,
                       int row0, int ld, int k0, int lane) {
    const int r = lane & 15, h = lane >> 4;
    const unsigned short* p = base + (size_t)(row0 + r) * (size_t)ld + (size_t)(k0 + h * 8);
    FragU f;
    f.q[0] = *(const uint4*)(p);
    f.q[1] = *(const uint4*)(p + 16);
    return f.v;
}

// B fragment, 32x16 bf16 (KxN); column n is a row of the row-major weight [N,K].
// ISA 7.12.2 (B layout): lanes 0-15 hold col=lane K chunks {0..7,16..23}; lanes 16-31 K chunks {8..15,24..31}.
DEVFN v16bf load_fragB(const unsigned short* __restrict__ base,
                       int col0, int ld, int k0, int lane) {
    const int r = lane & 15, h = lane >> 4;
    const unsigned short* p = base + (size_t)(col0 + r) * (size_t)ld + (size_t)(k0 + h * 8);
    FragU f;
    f.q[0] = *(const uint4*)(p);
    f.q[1] = *(const uint4*)(p + 16);
    return f.v;
}

// ---------------------------------------------------------------------------
// Device-side scalar coefficients (betas / T live in device memory):
//   coeff[t]     = c_t  = 1 - beta1^(t+1)                       t = 0..T-1
//   coeff[64+t]  = a_t  = (1-b2)(1-b3) * sum_{s>=t} b3^(T-1-s) b2^(s-t)
//   coeff[128]   = Sb   = 1 - beta3^T
// ---------------------------------------------------------------------------
__global__ void prep_coeffs_kernel(const float* __restrict__ b_taus,
                                   const int*  __restrict__ Tptr,
                                   float* __restrict__ coeff) {
    if (threadIdx.x == 0 && blockIdx.x == 0) {
        int T = *Tptr;
        if (T < 1)  T = 1;
        if (T > 64) T = 64;
        const float beta1 = 1.f / (1.f + expf(-b_taus[0]));
        const float beta2 = 1.f / (1.f + expf(-b_taus[1]));
        const float beta3 = 1.f / (1.f + expf(-b_taus[2]));
        float p1 = 1.f;
        for (int t = 0; t < 64; ++t) {
            if (t < T) { p1 *= beta1; coeff[t] = 1.f - p1; }
            else coeff[t] = 0.f;
        }
        for (int tau = 0; tau < 64; ++tau) {
            float s = 0.f;
            if (tau < T) {
                float p2 = 1.f;
                for (int t = tau; t < T; ++t) {
                    s += powf(beta3, (float)(T - 1 - t)) * p2;
                    p2 *= beta2;
                }
                s *= (1.f - beta2) * (1.f - beta3);
            }
            coeff[64 + tau] = s;
        }
        coeff[128] = 1.f - powf(beta3, (float)T);
    }
}

__global__ void cvt_f32_to_bf16_kernel(const float* __restrict__ in,
                                       unsigned short* __restrict__ out,
                                       size_t n) {
    size_t i = (size_t)blockIdx.x * blockDim.x + threadIdx.x;
    const size_t stride = (size_t)gridDim.x * blockDim.x;
    for (; i < n; i += stride) out[i] = f2bf(in[i]);
}

// ---------------------------------------------------------------------------
// NT GEMM:  out[m,n] = sum_k A[m,k] * W[n,k]   (A: [M,K] bf16, W: [N,K] bf16)
// Block: 256 threads = 8 waves (2x4).  Wave tile 64x64 = 4x4 WMMA tiles.
// MODE 0: out(bf16) = relu(acc + bias[n])                       (layer 1)
// MODE 1: out(bf16) = sum_t a[t]*relu(c[t]*acc + bias[n])       (layer 2 + recursion)
// MODE 2: out(f32)  = acc + Sb*bias[n]                          (layer 3)
// ---------------------------------------------------------------------------
template <int MODE>
__global__ __launch_bounds__(256)
void gemm_wmma_kernel(const unsigned short* __restrict__ A,
                      const unsigned short* __restrict__ W,
                      void* __restrict__ Out,
                      int M, int N, int K,
                      const float* __restrict__ bias,
                      const float* __restrict__ coeff,
                      const int*  __restrict__ Tptr) {
    const int lane  = threadIdx.x & 31;
    const int wave  = threadIdx.x >> 5;
    const int waveM = wave & 1;
    const int waveN = wave >> 1;
    const int m0 = blockIdx.y * 128 + waveM * 64;
    const int n0 = blockIdx.x * 256 + waveN * 64;

    __shared__ float s_c[64];
    __shared__ float s_a[64];
    int T = 0;
    if (MODE == 1) {
        T = *Tptr;
        if (T < 1)  T = 1;
        if (T > 64) T = 64;
        if (threadIdx.x < 64) {
            s_c[threadIdx.x] = coeff[threadIdx.x];
            s_a[threadIdx.x] = coeff[64 + threadIdx.x];
        }
        __syncthreads();
    }

    v8f acc[4][4];
    const v8f vzero = {0.f, 0.f, 0.f, 0.f, 0.f, 0.f, 0.f, 0.f};
#pragma unroll
    for (int mt = 0; mt < 4; ++mt)
#pragma unroll
        for (int nt = 0; nt < 4; ++nt) acc[mt][nt] = vzero;

    for (int k0 = 0; k0 < K; k0 += 32) {
        v16bf af[4], bfr[4];
#pragma unroll
        for (int mt = 0; mt < 4; ++mt) af[mt] = load_fragA(A, m0 + mt * 16, K, k0, lane);
#pragma unroll
        for (int nt = 0; nt < 4; ++nt) bfr[nt] = load_fragB(W, n0 + nt * 16, K, k0, lane);
#pragma unroll
        for (int mt = 0; mt < 4; ++mt) {
#pragma unroll
            for (int nt = 0; nt < 4; ++nt) {
                acc[mt][nt] = __builtin_amdgcn_wmma_f32_16x16x32_bf16(
                    /*neg_a=*/false, af[mt], /*neg_b=*/false, bfr[nt],
                    /*c_mod=*/(short)0, acc[mt][nt],
                    /*reuse_a=*/false, /*reuse_b=*/false);
            }
        }
    }

    // C/D layout: VGPR i -> (M=i, N=lane) for lanes 0-15, (M=8+i, N=lane-16) for 16-31.
    const int r = lane & 15, h = lane >> 4;
    const float Sb = (MODE == 2) ? coeff[128] : 0.f;
#pragma unroll
    for (int mt = 0; mt < 4; ++mt) {
#pragma unroll
        for (int nt = 0; nt < 4; ++nt) {
            const int col = n0 + nt * 16 + r;
            const float bcol = bias[col];
#pragma unroll
            for (int i = 0; i < 8; ++i) {
                const int row = m0 + mt * 16 + 8 * h + i;
                const float v = acc[mt][nt][i];
                if (MODE == 0) {
                    const float d = fmaxf(v + bcol, 0.f);
                    ((unsigned short*)Out)[(size_t)row * N + col] = f2bf(d);
                } else if (MODE == 1) {
                    float z = 0.f;
                    for (int t = 0; t < T; ++t)
                        z += s_a[t] * fmaxf(s_c[t] * v + bcol, 0.f);
                    ((unsigned short*)Out)[(size_t)row * N + col] = f2bf(z);
                } else {
                    ((float*)Out)[(size_t)row * N + col] = v + Sb * bcol;
                }
            }
        }
    }
}

// ---------------------------------------------------------------------------
extern "C" void kernel_launch(void* const* d_in, const int* in_sizes, int n_in,
                              void* d_out, int out_size, void* d_ws, size_t ws_size,
                              hipStream_t stream) {
    (void)in_sizes; (void)n_in; (void)out_size; (void)ws_size;

    constexpr int Bsz = 1024, DIN = 2048, H1 = 4096, H2 = 4096, DOUT = 1024;

    const float* x     = (const float*)d_in[0];
    const float* W1    = (const float*)d_in[1];
    const float* b1    = (const float*)d_in[2];
    const float* W2    = (const float*)d_in[3];
    const float* b2    = (const float*)d_in[4];
    const float* W3    = (const float*)d_in[5];
    const float* b3    = (const float*)d_in[6];
    const float* btaus = (const float*)d_in[7];
    const int*   Tp    = (const int*)d_in[8];

    char*  ws    = (char*)d_ws;
    float* coeff = (float*)ws;                         // 129 floats used
    size_t off   = 1024;
    auto alloc = [&](size_t bytes) -> char* {
        char* p = ws + off;
        off += (bytes + 255) & ~(size_t)255;
        return p;
    };
    unsigned short* xb  = (unsigned short*)alloc((size_t)Bsz * DIN * 2);
    unsigned short* w1b = (unsigned short*)alloc((size_t)H1  * DIN * 2);
    unsigned short* w2b = (unsigned short*)alloc((size_t)H2  * H1  * 2);
    unsigned short* w3b = (unsigned short*)alloc((size_t)DOUT* H2  * 2);
    unsigned short* d1b = (unsigned short*)alloc((size_t)Bsz * H1  * 2);
    unsigned short* zb  = (unsigned short*)alloc((size_t)Bsz * H2  * 2);

    prep_coeffs_kernel<<<1, 1, 0, stream>>>(btaus, Tp, coeff);

    cvt_f32_to_bf16_kernel<<<2048, 256, 0, stream>>>(x,  xb,  (size_t)Bsz * DIN);
    cvt_f32_to_bf16_kernel<<<2048, 256, 0, stream>>>(W1, w1b, (size_t)H1  * DIN);
    cvt_f32_to_bf16_kernel<<<2048, 256, 0, stream>>>(W2, w2b, (size_t)H2  * H1);
    cvt_f32_to_bf16_kernel<<<2048, 256, 0, stream>>>(W3, w3b, (size_t)DOUT* H2);

    // Layer 1: d1 = relu(x @ W1^T + b1)  -> bf16
    gemm_wmma_kernel<0><<<dim3(H1 / 256, Bsz / 128), 256, 0, stream>>>(
        xb, w1b, d1b, Bsz, H1, DIN, b1, coeff, Tp);

    // Layer 2 (+ folded time recursion): z = sum_t a_t*relu(c_t*(d1@W2^T) + b2) -> bf16
    gemm_wmma_kernel<1><<<dim3(H2 / 256, Bsz / 128), 256, 0, stream>>>(
        d1b, w2b, zb, Bsz, H2, H1, b2, coeff, Tp);

    // Layer 3: s3 = z @ W3^T + (1 - beta3^T)*b3 -> f32 output
    gemm_wmma_kernel<2><<<dim3(DOUT / 256, Bsz / 128), 256, 0, stream>>>(
        zb, w3b, d_out, Bsz, DOUT, H2, b3, coeff, Tp);
}
